// TemporalAttention_35940286333099
// MI455X (gfx1250) — matrix operand order repaired and checked by reading, not compiled, not run on hardware
//
#include <hip/hip_runtime.h>

// ---- problem constants ----
#define NB   8192
#define TT   16
#define CC   512
#define NH   8
#define DH   64
#define GB   4                 // batch elements per workgroup
#define NBLK (NB / GB)         // 2048 workgroups
#define NTHR 512               // 16 waves

typedef __attribute__((ext_vector_type(16))) __bf16 v16bf;
typedef __attribute__((ext_vector_type(8)))  float  v8f;

union Frag { unsigned u[8]; v16bf v; };

__device__ __forceinline__ unsigned short f2bf(float f) {
  unsigned u = __builtin_bit_cast(unsigned, f);
  u += 0x7fffu + ((u >> 16) & 1u);          // round-to-nearest-even
  return (unsigned short)(u >> 16);
}
__device__ __forceinline__ unsigned pk2(unsigned short lo, unsigned short hi) {
  return (unsigned)lo | ((unsigned)hi << 16);
}

// ---------------- prep: W (fp32 row-major [k][n]) -> bf16 transposed [n][k] ----------------
__global__ void prep_weights(const float* __restrict__ Wq, const float* __restrict__ Wk,
                             const float* __restrict__ Wv, const float* __restrict__ Wo,
                             unsigned short* __restrict__ ws) {
  int idx = blockIdx.x * 256 + threadIdx.x;          // 4*512*512 elements
  int mat = idx >> 18;
  int e   = idx & 262143;
  int n = e >> 9, k = e & 511;
  const float* W = (mat == 0) ? Wq : (mat == 1) ? Wk : (mat == 2) ? Wv : Wo;
  ws[(size_t)mat * 262144 + (size_t)n * 512 + k] = f2bf(W[(size_t)k * 512 + n]);
}

// ---------------- LDS layout ----------------
// strides chosen for (a) 16B alignment of fragment dword-groups (ds_load_b128)
// and (b) bank-conflict-free row access
#define XS   520         // bf16 row stride for x/q/k/v/attn_out (1040B)
#define RKS  72          // RKb bf16 row stride (144B)
#define RVS  40          // RVt bf16 row stride (80B)
#define PS   33          // P f32 row stride
#define SS   17          // sim f32 row stride
#define OFF_X   0                       // x_bf16 [64][520]  (reused as attn_out)
#define OFF_Q   66560                   // q [64][520]
#define OFF_K   133120                  // k [64][520]
#define OFF_V   199680                  // v [64][520]
#define OFF_SIM 266240                  // per-wave sim [16][16][17] f32  (17408)
#define OFF_P   283648                  // per-wave P   [16][16][33] f32  (33792)
#define OFF_RKB 317440                  // RKb [32][72] bf16 (r-major, d contig)  (4608)
#define OFF_RVT 322048                  // RVt [64][40] bf16 (d-major, r contig)  (5120)
#define SMEM_SZ 327168                  // <= 320 KB

__device__ __forceinline__ int kbA(int j, int lh) {  // 16-bit A 16x32 K-pair base
  return (j << 1) + (lh << 4);
}
__device__ __forceinline__ int kbB(int j, int lh) {  // 16-bit B 32x16 K-pair base
  return (j << 1) + (lh << 4);
}

__global__ void __launch_bounds__(NTHR, 1)
fused_attn(const float* __restrict__ x, const unsigned short* __restrict__ wsw,
           const float* __restrict__ bo, const float* __restrict__ relk,
           const float* __restrict__ relv, float* __restrict__ out) {
  extern __shared__ char smem[];
  const int tid  = threadIdx.x;
  const int lane = tid & 31;
  const int wave = tid >> 5;          // 0..15
  const int nl   = lane & 15;
  const int lh   = lane >> 4;
  const int b    = blockIdx.x;        // batch group

  // ---- stage x for GB batches (fp32 -> bf16 in LDS), zero rel-table LDS ----
  const float4* x4 = (const float4*)(x + (size_t)b * GB * TT * CC);
  for (int i = tid; i < GB * TT * CC / 4; i += NTHR) {
    float4 vv = x4[i];
    int base = i << 2;
    int row = base >> 9, col = base & 511;    // row 0..63
    char* p = smem + OFF_X + ((row * XS + col) << 1);
    *(unsigned*)p       = pk2(f2bf(vv.x), f2bf(vv.y));
    *(unsigned*)(p + 4) = pk2(f2bf(vv.z), f2bf(vv.w));
  }
  unsigned short* rkb = (unsigned short*)(smem + OFF_RKB);
  unsigned short* rvt = (unsigned short*)(smem + OFF_RVT);
  for (int i = tid; i < 32 * RKS; i += NTHR) rkb[i] = 0;
  for (int i = tid; i < 64 * RVS; i += NTHR) rvt[i] = 0;
  __syncthreads();
  // fill rel tables (index range actually used is 1..31, so r<32 suffices)
  for (int i = tid; i < 33 * 64; i += NTHR) {
    int r = i >> 6, d = i & 63;
    if (r < 32) {
      rkb[r * RKS + d] = f2bf(relk[i]);
      rvt[d * RVS + r] = f2bf(relv[i]);
    }
  }
  __syncthreads();

  // ---- phase 1: q,k,v = x @ W ; each job = one (mat, 16-col tile), 4 M-tiles share B ----
  for (int job = wave; job < 96; job += 16) {
    int mat = job % 3;                 // 0=q 1=k 2=v
    int n0  = job / 3;                 // 0..31
    int ncol = (n0 << 4) + nl;
    const unsigned* wt = (const unsigned*)(wsw + (size_t)mat * 262144);
    // prefetch this lane's whole weight column (1 KB) before streaming it
#pragma unroll
    for (int pf = 0; pf < 4; ++pf)
      __builtin_prefetch((const void*)(wt + (ncol * 256) + pf * 64), 0, 3);
    v8f acc[GB];
#pragma unroll
    for (int g = 0; g < GB; ++g) acc[g] = (v8f){0.f,0.f,0.f,0.f,0.f,0.f,0.f,0.f};
#pragma unroll 4
    for (int k0 = 0; k0 < 512; k0 += 32) {
      Frag bb;
#pragma unroll
      for (int j = 0; j < 8; ++j)
        bb.u[j] = wt[(ncol * 512 + k0 + kbB(j, lh)) >> 1];
#pragma unroll
      for (int g = 0; g < GB; ++g) {
        Frag a;
#pragma unroll
        for (int j = 0; j < 8; ++j)
          a.u[j] = *(const unsigned*)(smem + OFF_X +
                     (((g * 16 + nl) * XS + k0 + kbA(j, lh)) << 1));
        acc[g] = __builtin_amdgcn_wmma_f32_16x16x32_bf16(false, a.v, false, bb.v,
                                                         (short)0, acc[g], false, false);
      }
    }
    unsigned short* dst =
        (unsigned short*)(smem + (mat == 0 ? OFF_Q : mat == 1 ? OFF_K : OFF_V));
#pragma unroll
    for (int g = 0; g < GB; ++g)
#pragma unroll
      for (int i = 0; i < 8; ++i)
        dst[(g * 16 + i + (lh << 3)) * XS + ncol] = f2bf(acc[g][i]);
  }
  __syncthreads();

  // ---- phase 2: attention; 32 (g,h) blocks over 16 waves ----
  for (int task = wave; task < GB * NH; task += 16) {
    const int g  = task >> 3;
    const int h  = task & 7;
    const int rb = g * 16;             // row base of this batch element
    const int qb = h * DH;             // head column base
    Frag qf0, qf1;                     // q A-fragments, K-chunks 0 and 32
#pragma unroll
    for (int j = 0; j < 8; ++j) {
      int kb = kbA(j, lh);
      qf0.u[j] = *(const unsigned*)(smem + OFF_Q + (((rb + nl) * XS + qb + kb) << 1));
      qf1.u[j] = *(const unsigned*)(smem + OFF_Q + (((rb + nl) * XS + qb + 32 + kb) << 1));
    }
    // sim = q @ k^T   (K = d = 64 -> two WMMA)
    v8f sacc = {0.f,0.f,0.f,0.f,0.f,0.f,0.f,0.f};
#pragma unroll
    for (int c = 0; c < 2; ++c) {
      Frag kf;
#pragma unroll
      for (int j = 0; j < 8; ++j)
        kf.u[j] = *(const unsigned*)(smem + OFF_K +
                    (((rb + nl) * XS + qb + c * 32 + kbB(j, lh)) << 1));
      sacc = __builtin_amdgcn_wmma_f32_16x16x32_bf16(false, (c ? qf1.v : qf0.v), false, kf.v,
                                                     (short)0, sacc, false, false);
    }
    float* simw = (float*)(smem + OFF_SIM) + wave * (16 * SS);
#pragma unroll
    for (int i = 0; i < 8; ++i) simw[(i + (lh << 3)) * SS + nl] = sacc[i];
    // P[t,r] = q[t] . relK[r]   (r in 0..31 -> 2 N-tiles x 2 K-chunks)
    float* pw = (float*)(smem + OFF_P) + wave * (16 * PS);
#pragma unroll
    for (int nt = 0; nt < 2; ++nt) {
      v8f pacc = {0.f,0.f,0.f,0.f,0.f,0.f,0.f,0.f};
#pragma unroll
      for (int c = 0; c < 2; ++c) {
        Frag rbf;
#pragma unroll
        for (int j = 0; j < 8; ++j)
          rbf.u[j] = *(const unsigned*)(smem + OFF_RKB +
                       (((nt * 16 + nl) * RKS + c * 32 + kbB(j, lh)) << 1));
        pacc = __builtin_amdgcn_wmma_f32_16x16x32_bf16(false, (c ? qf1.v : qf0.v), false, rbf.v,
                                                       (short)0, pacc, false, false);
      }
#pragma unroll
      for (int i = 0; i < 8; ++i) pw[(i + (lh << 3)) * PS + nt * 16 + nl] = pacc[i];
    }
    // softmax rows (lanes 0..15 own row t=lane); rel-k gathered from P
    if (lane < 16) {
      const int t = lane;
      float* srow = simw + t * SS;
      float* prow = pw + t * PS;
      float vals[16];
      float m = -3.4e38f;
#pragma unroll
      for (int s = 0; s < 16; ++s) {
        float v2 = (srow[s] + prow[s - t + 16]) * 0.125f;   // scale = d^-0.5 = 1/8
        vals[s] = v2; m = fmaxf(m, v2);
      }
      float sum = 0.f;
#pragma unroll
      for (int s = 0; s < 16; ++s) { float e = __expf(vals[s] - m); vals[s] = e; sum += e; }
      float inv = 1.f / sum;
#pragma unroll
      for (int s = 0; s < 16; ++s) srow[s] = vals[s] * inv;
    }
    // attn A-fragment (K padded 16->32 with zeros) and banded A2 for rel-v
    const float* srow = simw + nl * SS;
    Frag af, a2f;
#pragma unroll
    for (int j = 0; j < 8; ++j) {
      int kb = kbA(j, lh);
      unsigned short lo = (kb     < 16) ? f2bf(srow[kb])     : (unsigned short)0;
      unsigned short hi = (kb + 1 < 16) ? f2bf(srow[kb + 1]) : (unsigned short)0;
      af.u[j] = pk2(lo, hi);
      int s0 = kb + nl - 16, s1 = s0 + 1;   // A2[t,r] = attn[t, r+t-16]
      lo = (s0 >= 0 && s0 < 16) ? f2bf(srow[s0]) : (unsigned short)0;
      hi = (s1 >= 0 && s1 < 16) ? f2bf(srow[s1]) : (unsigned short)0;
      a2f.u[j] = pk2(lo, hi);
    }
    // out = attn@v + A2@relV, four 16-wide d tiles; write to attn_out (reuses x LDS)
    unsigned short* ao = (unsigned short*)(smem + OFF_X);
    const unsigned short* vb = (const unsigned short*)(smem + OFF_V);
#pragma unroll
    for (int dt = 0; dt < 4; ++dt) {
      v8f oacc = {0.f,0.f,0.f,0.f,0.f,0.f,0.f,0.f};
      Frag bv, br;
      int dcol = qb + dt * 16 + nl;
#pragma unroll
      for (int j = 0; j < 8; ++j) {
        int kb2 = kbB(j, lh);
        // v rows rb+kb2, rb+kb2+1 (K = s); K>=16 is zero padding (lanes 16..31)
        bv.u[j] = lh ? 0u
                     : pk2(vb[(rb + kb2) * XS + dcol], vb[(rb + kb2 + 1) * XS + dcol]);
        br.u[j] = *(const unsigned*)(smem + OFF_RVT +
                    (((dt * 16 + nl) * RVS + kb2) << 1));
      }
      oacc = __builtin_amdgcn_wmma_f32_16x16x32_bf16(false, af.v, false, bv.v,
                                                     (short)0, oacc, false, false);
      oacc = __builtin_amdgcn_wmma_f32_16x16x32_bf16(false, a2f.v, false, br.v,
                                                     (short)0, oacc, false, false);
#pragma unroll
      for (int i = 0; i < 8; ++i)
        ao[(rb + i + (lh << 3)) * XS + qb + dt * 16 + nl] = f2bf(oacc[i]);
    }
  }
  __syncthreads();

  // ---- phase 3: out = attn_out @ Wo + bo ; 32 column tiles, 4 M-tiles share B ----
  const unsigned* wto = (const unsigned*)(wsw + (size_t)3 * 262144);
  for (int n0 = wave; n0 < 32; n0 += 16) {
    int ncol = (n0 << 4) + nl;
#pragma unroll
    for (int pf = 0; pf < 4; ++pf)
      __builtin_prefetch((const void*)(wto + (ncol * 256) + pf * 64), 0, 3);
    v8f acc[GB];
#pragma unroll
    for (int g = 0; g < GB; ++g) acc[g] = (v8f){0.f,0.f,0.f,0.f,0.f,0.f,0.f,0.f};
#pragma unroll 4
    for (int k0 = 0; k0 < 512; k0 += 32) {
      Frag bb;
#pragma unroll
      for (int j = 0; j < 8; ++j)
        bb.u[j] = wto[(ncol * 512 + k0 + kbB(j, lh)) >> 1];
#pragma unroll
      for (int g = 0; g < GB; ++g) {
        Frag a;
#pragma unroll
        for (int j = 0; j < 8; ++j)
          a.u[j] = *(const unsigned*)(smem + OFF_X +
                     (((g * 16 + nl) * XS + k0 + kbA(j, lh)) << 1));
        acc[g] = __builtin_amdgcn_wmma_f32_16x16x32_bf16(false, a.v, false, bb.v,
                                                         (short)0, acc[g], false, false);
      }
    }
    float bov = bo[ncol];
#pragma unroll
    for (int g = 0; g < GB; ++g)
#pragma unroll
      for (int i = 0; i < 8; ++i) {
        int row = i + (lh << 3);
        out[((size_t)(b * GB + g) * TT + row) * CC + ncol] = acc[g][i] + bov;
      }
  }
}

extern "C" void kernel_launch(void* const* d_in, const int* in_sizes, int n_in,
                              void* d_out, int out_size, void* d_ws, size_t ws_size,
                              hipStream_t stream) {
  (void)in_sizes; (void)n_in; (void)out_size; (void)ws_size;
  const float* x  = (const float*)d_in[0];
  const float* Wq = (const float*)d_in[1];
  const float* Wk = (const float*)d_in[2];
  const float* Wv = (const float*)d_in[3];
  const float* Wo = (const float*)d_in[4];
  const float* bo = (const float*)d_in[5];
  const float* rk = (const float*)d_in[6];
  const float* rv = (const float*)d_in[7];
  unsigned short* ws = (unsigned short*)d_ws;   // 2 MB of bf16 transposed weights

  prep_weights<<<4096, 256, 0, stream>>>(Wq, Wk, Wv, Wo, ws);
  fused_attn<<<NBLK, NTHR, SMEM_SZ, stream>>>(x, ws, bo, rk, rv, (float*)d_out);
}
